// VGAE_56195352101194
// MI455X (gfx1250) — compile-verified, matrix operand order checked by reading, not run on hardware
//
#include <hip/hip_runtime.h>
#include <hip/hip_bf16.h>

typedef __bf16 bf16_t;
typedef __attribute__((ext_vector_type(8)))  bf16_t v8bf;
typedef __attribute__((ext_vector_type(16))) bf16_t v16bf;
typedef __attribute__((ext_vector_type(8)))  float  v8f;

// ---------------------------------------------------------------- utilities
__global__ void k_zero_f32(float* __restrict__ p, long n) {
  long i = (long)blockIdx.x * blockDim.x + threadIdx.x;
  if (i < n) p[i] = 0.0f;
}

__global__ void k_fill_f32(float* __restrict__ p, int n, float v) {
  int i = blockIdx.x * blockDim.x + threadIdx.x;
  if (i < n) p[i] = v;
}

// W [K, Ncol] row-major (fp32)  ->  Wt [Ncol, K] row-major (bf16)
__global__ void k_transpose_bf16(const float* __restrict__ W, bf16_t* __restrict__ Wt,
                                 int K, int Ncol) {
  int i = blockIdx.x * blockDim.x + threadIdx.x;
  if (i >= K * Ncol) return;
  int n = i / K, k = i - n * K;
  Wt[i] = (bf16_t)W[k * Ncol + n];
}

// Concatenate Wmu[K,O] and Wls[K,O] into Wt[(2O), K] bf16 (transposed, n-major)
__global__ void k_wcat_bf16(const float* __restrict__ Wmu, const float* __restrict__ Wls,
                            bf16_t* __restrict__ Wt, int K, int O) {
  int i = blockIdx.x * blockDim.x + threadIdx.x;
  if (i >= 2 * O * K) return;
  int n = i / K, k = i - n * K;
  float v = (n < O) ? Wmu[k * O + n] : Wls[k * O + (n - O)];
  Wt[i] = (bf16_t)v;
}

// ------------------------------------------- EmbeddingBag(sum) + L2 normalize
// one block per node, blockDim = D threads (D=128)
__global__ void k_embed(const int* __restrict__ fidx, const int* __restrict__ foff,
                        const float* __restrict__ fw, const float* __restrict__ emb,
                        bf16_t* __restrict__ xb, int N, int D, int total) {
  extern __shared__ float sm[];
  int n = blockIdx.x;
  int d = threadIdx.x;
  int s = foff[n];
  int e = (n + 1 < N) ? foff[n + 1] : total;
  float acc = 0.0f;
  for (int j = s; j < e; ++j) {
    acc = fmaf(fw[j], emb[(size_t)fidx[j] * D + d], acc);
  }
  sm[d] = acc * acc;
  __syncthreads();
  for (int t = blockDim.x >> 1; t > 0; t >>= 1) {
    if (d < t) sm[d] += sm[d + t];
    __syncthreads();
  }
  float inv = 1.0f / fmaxf(sqrtf(sm[0]), 1e-12f);
  xb[(size_t)n * D + d] = (bf16_t)(acc * inv);
}

// ---------------------------------------------------------------- GCN norms
__global__ void k_deg_edges(const int* __restrict__ dst, float* __restrict__ deg, int E) {
  int e = blockIdx.x * blockDim.x + threadIdx.x;
  if (e < E) unsafeAtomicAdd(&deg[dst[e]], 1.0f);
}

__global__ void k_deg_fin(const float* __restrict__ deg, float* __restrict__ disq,
                          float* __restrict__ dinv, int N) {
  int i = blockIdx.x * blockDim.x + threadIdx.x;
  if (i >= N) return;
  float d = deg[i];
  disq[i] = rsqrtf(d);
  dinv[i] = 1.0f / d;
}

__global__ void k_edge_norm(const int* __restrict__ src, const int* __restrict__ dst,
                            const float* __restrict__ disq, float* __restrict__ enorm, int E) {
  int e = blockIdx.x * blockDim.x + threadIdx.x;
  if (e < E) enorm[e] = disq[src[e]] * disq[dst[e]];
}

// ---------------------------------------------------------------- WMMA GEMM
// C[M,Ncol] (fp32) = A[M,K] (bf16 row-major) @ Bt[Ncol,K]^T (bf16, n-major rows)
// one wave per 16x16 output tile; K multiple of 32; M,Ncol multiples of 16.
__global__ void k_gemm_wmma(const bf16_t* __restrict__ A, const bf16_t* __restrict__ Bt,
                            float* __restrict__ C, int M, int K, int Ncol) {
  int tileN = blockIdx.x * 16;
  int tileM = blockIdx.y * 16;
  int lane  = threadIdx.x;        // 0..31, wave32
  int half  = lane >> 4;          // 0: K 0..7/16..23, 1: K 8..15/24..31
  int l     = lane & 15;

  const bf16_t* arow = A  + (size_t)(tileM + l) * K + half * 8;
  const bf16_t* brow = Bt + (size_t)(tileN + l) * K + half * 8;

  v8f acc = {};
  for (int k0 = 0; k0 < K; k0 += 32) {
    v8bf a0 = *(const v8bf*)(arow + k0);
    v8bf a1 = *(const v8bf*)(arow + k0 + 16);
    v8bf b0 = *(const v8bf*)(brow + k0);
    v8bf b1 = *(const v8bf*)(brow + k0 + 16);
    v16bf af = __builtin_shufflevector(a0, a1, 0,1,2,3,4,5,6,7,8,9,10,11,12,13,14,15);
    v16bf bf = __builtin_shufflevector(b0, b1, 0,1,2,3,4,5,6,7,8,9,10,11,12,13,14,15);
    acc = __builtin_amdgcn_wmma_f32_16x16x32_bf16(
        /*neg_a=*/false, af, /*neg_b=*/false, bf,
        /*c_mod=*/(short)0, acc, /*reuse_a=*/false, /*reuse_b=*/false);
  }

  // C/D layout: lanes 0-15 -> rows 0..7, lanes 16-31 -> rows 8..15
  float* crow = C + (size_t)(tileM + half * 8) * Ncol + tileN + l;
  #pragma unroll
  for (int r = 0; r < 8; ++r) crow[(size_t)r * Ncol] = acc[r];
}

// --------------------------------------------------- edge scatter-add (dim=128)
// one wave per edge, 4 floats per lane, hardware fp32 atomics
__global__ void k_scatter(const int* __restrict__ src, const int* __restrict__ dst,
                          const float* __restrict__ enorm, const float* __restrict__ xw,
                          float* __restrict__ agg, int E, int dim) {
  int gid  = blockIdx.x * blockDim.x + threadIdx.x;
  int e    = gid >> 5;
  int lane = gid & 31;
  if (e >= E) return;
  int s = src[e], d = dst[e];
  float w = enorm[e];
  float4 v = *(const float4*)(xw + (size_t)s * dim + lane * 4);
  float* ap = agg + (size_t)d * dim + lane * 4;
  unsafeAtomicAdd(ap + 0, w * v.x);
  unsafeAtomicAdd(ap + 1, w * v.y);
  unsafeAtomicAdd(ap + 2, w * v.z);
  unsafeAtomicAdd(ap + 3, w * v.w);
}

// ---------------------------------------------------------- layer finalizers
__global__ void k_h_fin(const float* __restrict__ agg, const float* __restrict__ xw,
                        const float* __restrict__ dinv, const float* __restrict__ b,
                        bf16_t* __restrict__ hb, int N, int H) {
  long i = (long)blockIdx.x * blockDim.x + threadIdx.x;
  if (i >= (long)N * H) return;
  int n = (int)(i / H), c = (int)(i - (long)n * H);
  float v = agg[i] + xw[i] * dinv[n] + b[c];
  hb[i] = (bf16_t)fmaxf(v, 0.0f);
}

__global__ void k_z_fin(const float* __restrict__ aggm, const float* __restrict__ hm,
                        const float* __restrict__ dinv, const float* __restrict__ bmu,
                        const float* __restrict__ bls, const float* __restrict__ noise,
                        float* __restrict__ z, int N, int O) {
  long i = (long)blockIdx.x * blockDim.x + threadIdx.x;
  if (i >= (long)N * O) return;
  int n = (int)(i / O), c = (int)(i - (long)n * O);
  size_t base = (size_t)n * (2 * O);
  float di = dinv[n];
  float mu = aggm[base + c]     + hm[base + c]     * di + bmu[c];
  float ls = aggm[base + O + c] + hm[base + O + c] * di + bls[c];
  z[i] = mu + noise[i] * expf(ls);
}

// --------------------------------------------------------------- decoder
__global__ void k_decode(const int* __restrict__ src, const int* __restrict__ dst,
                         const float* __restrict__ z, float* __restrict__ out,
                         int E, int O) {
  int e = blockIdx.x * blockDim.x + threadIdx.x;
  if (e >= E) return;
  const float4* a = (const float4*)(z + (size_t)src[e] * O);
  const float4* b = (const float4*)(z + (size_t)dst[e] * O);
  float acc = 0.0f;
  for (int i = 0; i < O / 4; ++i) {
    float4 x = a[i], y = b[i];
    acc = fmaf(x.x, y.x, acc);
    acc = fmaf(x.y, y.y, acc);
    acc = fmaf(x.z, y.z, acc);
    acc = fmaf(x.w, y.w, acc);
  }
  out[e] = acc;
}

// ================================================================= launcher
extern "C" void kernel_launch(void* const* d_in, const int* in_sizes, int n_in,
                              void* d_out, int out_size, void* d_ws, size_t ws_size,
                              hipStream_t stream) {
  const int*   fidx  = (const int*)  d_in[0];
  const int*   foff  = (const int*)  d_in[1];
  const float* fw    = (const float*)d_in[2];
  const int*   eidx  = (const int*)  d_in[3];
  const float* noise = (const float*)d_in[4];
  const float* emb   = (const float*)d_in[5];
  const float* W1    = (const float*)d_in[6];
  const float* b1    = (const float*)d_in[7];
  const float* Wmu   = (const float*)d_in[8];
  const float* bmu   = (const float*)d_in[9];
  const float* Wls   = (const float*)d_in[10];
  const float* bls   = (const float*)d_in[11];

  const int N     = in_sizes[1];
  const int total = in_sizes[0];
  const int E     = in_sizes[3] / 2;
  const int H     = in_sizes[7];          // hidden dim (128)
  const int D     = in_sizes[6] / H;      // embedding dim (128)
  const int O     = in_sizes[9];          // out dim (64)
  const int O2    = 2 * O;                // fused mu|logstd width (128)

  const int* src = eidx;
  const int* dst = eidx + E;

  // workspace carve-out (256B aligned slices)
  char* wsp = (char*)d_ws;
  size_t off = 0;
  auto take = [&](size_t bytes) -> char* {
    char* p = wsp + off;
    off = (off + bytes + 255) & ~(size_t)255;
    return p;
  };
  bf16_t* xb    = (bf16_t*)take((size_t)N * D * 2);   // normalized node features, bf16
  bf16_t* w1t   = (bf16_t*)take((size_t)H * D * 2);   // W1^T bf16
  bf16_t* wcat  = (bf16_t*)take((size_t)O2 * H * 2);  // [Wmu|Wls]^T bf16
  float*  xw1   = (float*) take((size_t)N * H * 4);   // x @ W1
  float*  agg1  = (float*) take((size_t)N * H * 4);
  bf16_t* hb    = (bf16_t*)take((size_t)N * H * 2);   // relu hidden, bf16
  float*  hmuls = (float*) take((size_t)N * O2 * 4);  // h @ [Wmu|Wls]
  float*  aggm  = (float*) take((size_t)N * O2 * 4);
  float*  z     = (float*) take((size_t)N * O * 4);
  float*  deg   = (float*) take((size_t)N * 4);
  float*  disq  = (float*) take((size_t)N * 4);
  float*  dinv  = (float*) take((size_t)N * 4);
  float*  enorm = (float*) take((size_t)E * 4);

  // --- weight prep (tiny) ---
  k_transpose_bf16<<<(H * D + 255) / 256, 256, 0, stream>>>(W1, w1t, D, H);
  k_wcat_bf16<<<(O2 * H + 255) / 256, 256, 0, stream>>>(Wmu, Wls, wcat, H, O);

  // --- embedding bag + L2 normalize ---
  k_embed<<<N, D, D * (int)sizeof(float), stream>>>(fidx, foff, fw, emb, xb, N, D, total);

  // --- GCN normalization terms ---
  k_fill_f32<<<(N + 255) / 256, 256, 0, stream>>>(deg, N, 1.0f);   // self loop
  k_deg_edges<<<(E + 255) / 256, 256, 0, stream>>>(dst, deg, E);
  k_deg_fin<<<(N + 255) / 256, 256, 0, stream>>>(deg, disq, dinv, N);
  k_edge_norm<<<(E + 255) / 256, 256, 0, stream>>>(src, dst, disq, enorm, E);

  // --- layer 1: xw1 = x @ W1 ; agg over edges ; relu ---
  k_zero_f32<<<(int)(((long)N * H + 255) / 256), 256, 0, stream>>>(agg1, (long)N * H);
  {
    dim3 g(H / 16, N / 16);
    k_gemm_wmma<<<g, 32, 0, stream>>>(xb, w1t, xw1, N, D, H);
  }
  k_scatter<<<(int)(((long)E * 32 + 255) / 256), 256, 0, stream>>>(src, dst, enorm, xw1, agg1, E, H);
  k_h_fin<<<(int)(((long)N * H + 255) / 256), 256, 0, stream>>>(agg1, xw1, dinv, b1, hb, N, H);

  // --- fused mu/logstd layer: hmuls = h @ [Wmu|Wls] ; agg ; z ---
  k_zero_f32<<<(int)(((long)N * O2 + 255) / 256), 256, 0, stream>>>(aggm, (long)N * O2);
  {
    dim3 g(O2 / 16, N / 16);
    k_gemm_wmma<<<g, 32, 0, stream>>>(hb, wcat, hmuls, N, H, O2);
  }
  k_scatter<<<(int)(((long)E * 32 + 255) / 256), 256, 0, stream>>>(src, dst, enorm, hmuls, aggm, E, O2);
  k_z_fin<<<(int)(((long)N * O + 255) / 256), 256, 0, stream>>>(aggm, hmuls, dinv, bmu, bls, noise, z, N, O);

  // --- decoder: per-edge dot product ---
  k_decode<<<(E + 255) / 256, 256, 0, stream>>>(src, dst, z, (float*)d_out, E, O);
}
